// LSTMWithAdaptersWithPrameterGenerator_9552007266507
// MI455X (gfx1250) — matrix operand order repaired and checked by reading, not compiled, not run on hardware
//
#include <hip/hip_runtime.h>

// ---------------------------------------------------------------------------
// MI455X (gfx1250) implementation of the hypernet-adapter BiLSTM.
//
// - All GEMMs: bf16 WMMA (v_wmma_f32_16x16x32_bf16, fp32 accum).
// - LDS tile fills use CDNA5 async path (global_load_async_to_lds_b128,
//   ASYNCcnt) instead of VGPR round-trips.
// - The sequential recurrence is a persistent 64-block kernel; each block
//   pins its 128x1024 w_hh slice (256KB bf16) in the WGP's 320KB LDS for all
//   512 steps, loaded once via the Tensor Data Mover (tensor_load_to_lds,
//   TENSORcnt).  Steps are separated by a software grid barrier.
// - Total HBM traffic ~0.5GB << 23.3TB/s; weights fit in 192MB L2, so the
//   design minimizes the recurrence critical path, not bandwidth.
// ---------------------------------------------------------------------------

typedef __attribute__((ext_vector_type(16))) __bf16 v16bf;
typedef __attribute__((ext_vector_type(8)))  __bf16 v8bf;
typedef __attribute__((ext_vector_type(8)))  float  v8f;
typedef __attribute__((ext_vector_type(4)))  unsigned u32x4;
typedef __attribute__((ext_vector_type(8)))  unsigned u32x8;

#define HDIM       1024
#define DMODEL     2048
#define DADAPT     64
#define WSZ        (DMODEL * DADAPT)        // 131072
#define LAYERSZ    (2 * WSZ)                // 262144
#define BB         8
#define TT         512
#define MROWS      (BB * TT)                // 4096
#define NGATE      (4 * HDIM)               // 4096

// ---------------- CDNA5 async / TDM helpers --------------------------------
// Generic LDS pointer -> 32-bit LDS byte address (flat aperture maps addr[31:0]).
__device__ __forceinline__ unsigned lds_off(const void* p) {
    return (unsigned)(size_t)p;
}
// 16B global -> LDS async copy (ASYNCcnt-tracked, no VGPR data round-trip).
__device__ __forceinline__ void async_ld16(unsigned lds_byte, const void* g) {
    asm volatile("global_load_async_to_lds_b128 %0, %1, off"
                 :: "v"(lds_byte), "v"((unsigned long long)(size_t)g)
                 : "memory");
}
__device__ __forceinline__ void wait_async() {
    asm volatile("s_wait_asynccnt 0x0" ::: "memory");
}
// TDM: 2-D tile (tile_rows x width bf16, row stride row_stride elems) -> LDS.
// D# groups built in SGPRs per ISA 8.3/8.4 (count=1, data_size=2B, type=2).
__device__ __forceinline__ void tdm_load_2d(unsigned lds_byte, const void* gaddr,
                                            unsigned width, unsigned rows,
                                            unsigned row_stride, unsigned tile_rows) {
    unsigned long long ga = (unsigned long long)(size_t)gaddr;
    u32x4 g0;
    g0[0] = 1u;                                           // count=1 (user D#)
    g0[1] = lds_byte;                                     // lds_addr
    g0[2] = (unsigned)(ga & 0xffffffffu);                 // global_addr lo
    g0[3] = (unsigned)((ga >> 32) & 0x1ffffffu) | 0x80000000u; // hi25 | type=2
    u32x8 g1;
    g1[0] = 1u << 16;                                     // data_size=1 (2B)
    g1[1] = (width & 0xffffu) << 16;                      // tensor_dim0 lo16
    g1[2] = ((width >> 16) & 0xffffu) | ((rows & 0xffffu) << 16); // dim0 hi|dim1 lo
    g1[3] = ((rows >> 16) & 0xffffu) | ((width & 0xffffu) << 16); // dim1 hi|tile_dim0
    g1[4] = tile_rows & 0xffffu;                          // tile_dim1 (dim2=0)
    g1[5] = row_stride;                                   // dim0_stride lo32
    g1[6] = 0u;
    g1[7] = 0u;
    asm volatile("tensor_load_to_lds %0, %1" :: "s"(g0), "s"(g1) : "memory");
}

// ---------------- WMMA fragment loaders (ISA 7.12.2 layouts) ---------------
__device__ __forceinline__ v16bf frag_a(const __bf16* base, int ldk, int kb, int lane) {
    int m = lane & 15, hi = lane >> 4;
    const v8bf* p0 = (const v8bf*)(base + (size_t)m * ldk + kb + 8 * hi);
    const v8bf* p1 = (const v8bf*)(base + (size_t)m * ldk + kb + 16 + 8 * hi);
    return __builtin_shufflevector(*p0, *p1, 0,1,2,3,4,5,6,7,8,9,10,11,12,13,14,15);
}
__device__ __forceinline__ v16bf frag_b(const __bf16* base, int ldk, int kb, int lane) {
    int n = lane & 15, hi = lane >> 4;
    const v8bf* p0 = (const v8bf*)(base + (size_t)n * ldk + kb + 8 * hi);
    const v8bf* p1 = (const v8bf*)(base + (size_t)n * ldk + kb + 16 + 8 * hi);
    return __builtin_shufflevector(*p0, *p1, 0,1,2,3,4,5,6,7,8,9,10,11,12,13,14,15);
}
__device__ __forceinline__ v8f wmma_bf16(v16bf a, v16bf b, v8f c) {
    return __builtin_amdgcn_wmma_f32_16x16x32_bf16(false, a, false, b, (short)0, c, false, false);
}

// ---------------- small utility kernels ------------------------------------
__global__ void k_params(const float* __restrict__ hyper_w,
                         const float* __restrict__ lang_emb,
                         const int*   __restrict__ lang_id,
                         float* __restrict__ params, int n) {
    __shared__ float e[32];
    if (threadIdx.x < 32) e[threadIdx.x] = lang_emb[(*lang_id) * 32 + threadIdx.x];
    __syncthreads();
    int i = blockIdx.x * blockDim.x + threadIdx.x;
    if (i < n) {
        const float* w = hyper_w + (size_t)i * 32;
        float s = 0.f;
#pragma unroll
        for (int k = 0; k < 32; ++k) s += w[k] * e[k];
        params[i] = s;
    }
}

__global__ void k_f32_to_bf16(const float* __restrict__ in, __bf16* __restrict__ out,
                              long long n) {
    long long i = (long long)blockIdx.x * blockDim.x + threadIdx.x;
    long long stride = (long long)gridDim.x * blockDim.x;
    for (; i < n; i += stride) out[i] = (__bf16)in[i];
}

__global__ void k_zero_u32(unsigned* p, int n) {
    int i = blockIdx.x * blockDim.x + threadIdx.x;
    if (i < n) p[i] = 0u;
}

// ---------------- generic NT GEMM: C[M,N] = A[M,K] * B[N,K]^T ---------------
// block = 256 threads = 8 waves, tile 64(M) x 128(N) x 32(K).
// LDS staging uses async-to-LDS (ASYNCcnt) -- no VGPR round trip.
__global__ void k_gemm_nt(const __bf16* __restrict__ A, const __bf16* __restrict__ Bm,
                          float* __restrict__ C,
                          const float* __restrict__ bias0, const float* __restrict__ bias1,
                          int M, int N, int K, int relu) {
    __shared__ __bf16 As[64][40];     // 32 K used, padded stride
    __shared__ __bf16 Bs[128][40];
    const int tid  = threadIdx.x;
    const int lane = tid & 31, wid = tid >> 5;
    const int mw = wid & 3;           // 4 M-subtiles of 16
    const int nw = wid >> 2;          // 2 N-groups of 64
    const int m0 = blockIdx.y * 64, n0 = blockIdx.x * 128;
    v8f acc[4] = {};

    for (int k0 = 0; k0 < K; k0 += 32) {
        __syncthreads();              // WAR on tiles (implies dscnt drain)
        {   // A tile: 64 rows x 32 K, one 16B async chunk per thread
            int r = tid >> 2, cc = (tid & 3) * 8;
            int gm = m0 + r;
            if (gm < M) async_ld16(lds_off(&As[r][cc]), A + (size_t)gm * K + k0 + cc);
            else        { v8bf z = {}; *(v8bf*)&As[r][cc] = z; }
        }
#pragma unroll
        for (int i = 0; i < 2; ++i) {  // B tile: 128 rows x 32 K
            int idx = tid + i * 256;
            int r = idx >> 2, cc = (idx & 3) * 8;
            int gn = n0 + r;
            if (gn < N) async_ld16(lds_off(&Bs[r][cc]), Bm + (size_t)gn * K + k0 + cc);
            else        { v8bf z = {}; *(v8bf*)&Bs[r][cc] = z; }
        }
        if (k0 + 32 < K) {            // hint next tiles toward L2
            __builtin_prefetch(A  + (size_t)(m0 + (tid >> 2)) * K + k0 + 32, 0, 1);
            __builtin_prefetch(Bm + (size_t)(n0 + (tid >> 2)) * K + k0 + 32, 0, 1);
        }
        wait_async();
        __syncthreads();
        v16bf a = frag_a(&As[mw * 16][0], 40, 0, lane);
#pragma unroll
        for (int ti = 0; ti < 4; ++ti) {
            v16bf b = frag_b(&Bs[nw * 64 + ti * 16][0], 40, 0, lane);
            acc[ti] = wmma_bf16(a, b, acc[ti]);
        }
    }

    const int hi = lane >> 4, col = lane & 15;
#pragma unroll
    for (int ti = 0; ti < 4; ++ti) {
#pragma unroll
        for (int v = 0; v < 8; ++v) {
            int m = m0 + mw * 16 + v + 8 * hi;
            int n = n0 + nw * 64 + ti * 16 + col;
            if (m < M && n < N) {
                float val = acc[ti][v];
                if (bias0) val += bias0[n];
                if (bias1) val += bias1[n];
                if (relu)  val = fmaxf(val, 0.f);
                C[(size_t)m * N + n] = val;
            }
        }
    }
}

// ---------------- grid barrier (monotonic counter) --------------------------
__device__ __forceinline__ void grid_sync(unsigned* cnt, unsigned nb, unsigned& phase, int tid) {
    __syncthreads();
    if (tid == 0) {
        phase += nb;
        __hip_atomic_fetch_add(cnt, 1u, __ATOMIC_ACQ_REL, __HIP_MEMORY_SCOPE_AGENT);
        while (__hip_atomic_load(cnt, __ATOMIC_ACQUIRE, __HIP_MEMORY_SCOPE_AGENT) < phase)
            __builtin_amdgcn_s_sleep(1);
    }
    __syncthreads();
}

// ---------------- persistent BiLSTM recurrence ------------------------------
// 64 blocks: dir = blk/32, j-slice of 32 hidden units = blk%32.
// Dynamic LDS: w_hh slice 128x1024 bf16 (256KB, TDM-loaded once, resident all
// 512 steps), h (16x1024 bf16, batch padded), gates, cell state.
__global__ void k_recurrence(const float* __restrict__ xgF, const float* __restrict__ xgR,
                             const __bf16* __restrict__ whhF, const __bf16* __restrict__ whhR,
                             float* __restrict__ y,
                             __bf16* __restrict__ hbufF, __bf16* __restrict__ hbufR,
                             float* __restrict__ h_out, float* __restrict__ c_out,
                             unsigned* __restrict__ cnt) {
    extern __shared__ __align__(16) unsigned char smem[];
    __bf16* Wlds = (__bf16*)smem;                                   // 128*1024
    __bf16* Hlds = (__bf16*)(smem + 128 * 1024 * 2);                // 16*1024
    float*  Glds = (float*)(smem + 128 * 1024 * 2 + 16 * 1024 * 2); // 128*8
    float*  Clds = Glds + 128 * 8;                                  // 32*8

    const int tid  = threadIdx.x;
    const int lane = tid & 31, wid = tid >> 5;
    const int dir  = blockIdx.x >> 5;
    const int jbase = (blockIdx.x & 31) * 32;
    const float*  xg   = dir ? xgR : xgF;
    const __bf16* whh  = dir ? whhR : whhF;
    __bf16*       hbuf = dir ? hbufR : hbufF;
    const unsigned NB = gridDim.x;

    // TDM preload: 4 tiles (one per gate), each 32 rows x 1024 bf16, row
    // stride 1024, landing contiguously at Wlds rows gate*32..gate*32+31.
    if (wid == 0) {
#pragma unroll
        for (int g = 0; g < 4; ++g) {
            const __bf16* src = whh + ((size_t)(g * HDIM + jbase)) * HDIM;
            tdm_load_2d(lds_off(Wlds) + (unsigned)g * 32u * HDIM * 2u,
                        src, HDIM, 32u, HDIM, 32u);
        }
        __builtin_amdgcn_s_wait_tensorcnt(0);
    }
    for (int idx = tid; idx < 16 * 1024; idx += 256) Hlds[idx] = (__bf16)0.0f;
    for (int idx = tid; idx < 32 * 8;   idx += 256) Clds[idx] = 0.f;
    __syncthreads();

    unsigned phase = 0;
    for (int s = 0; s < TT; ++s) {
        const int tt = dir ? (TT - 1 - s) : s;

        // gates[16 x 128] = Hlds(16x1024) @ Wlds(128x1024)^T, one 16-N tile/wave
        v8f acc = {};
        const __bf16* wbase = Wlds + (size_t)(wid * 16) * 1024;
#pragma unroll 4
        for (int kb = 0; kb < HDIM; kb += 32) {
            v16bf a = frag_a(Hlds, 1024, kb, lane);
            v16bf b = frag_b(wbase, 1024, kb, lane);
            acc = wmma_bf16(a, b, acc);
        }
        {   // scatter to Glds[n_local][batch] with xg added
            const int col = lane & 15, hi = lane >> 4;
            const int nl  = wid * 16 + col;
            const int ngl = ((nl >> 5) << 10) + jbase + (nl & 31);
#pragma unroll
            for (int v = 0; v < 8; ++v) {
                int b = v + 8 * hi;
                if (b < 8)
                    Glds[nl * 8 + b] = acc[v] + xg[((size_t)b * TT + tt) * NGATE + ngl];
            }
        }
        __syncthreads();

        {   // gate fusion: 256 threads = 32 j x 8 batch
            const int b = tid & 7, jl = tid >> 3;
            float gi = Glds[(jl)      * 8 + b];
            float gf = Glds[(32 + jl) * 8 + b];
            float gg = Glds[(64 + jl) * 8 + b];
            float go = Glds[(96 + jl) * 8 + b];
            gi = 1.f / (1.f + __expf(-gi));
            gf = 1.f / (1.f + __expf(-gf));
            gg = tanhf(gg);
            go = 1.f / (1.f + __expf(-go));
            float c = gf * Clds[jl * 8 + b] + gi * gg;
            Clds[jl * 8 + b] = c;
            float h = go * tanhf(c);
            const int j = jbase + jl;
            y[((size_t)b * TT + tt) * DMODEL + dir * HDIM + j] = h;
            hbuf[b * HDIM + j] = (__bf16)h;
            if (s == TT - 1 && h_out) {
                h_out[dir * (BB * HDIM) + b * HDIM + j] = h;
                c_out[dir * (BB * HDIM) + b * HDIM + j] = c;
            }
        }
        __threadfence();
        grid_sync(cnt, NB, phase, tid);     // all h written

        // refresh Hlds rows 0..7 from global h: async 16B copies
        for (int idx = tid * 8; idx < 8 * HDIM; idx += 256 * 8)
            async_ld16(lds_off(Hlds + idx), hbuf + idx);
        wait_async();
        grid_sync(cnt, NB, phase, tid);     // all h read before next overwrite
    }
}

// ---------------- LayerNorm (block per row, in-place safe) ------------------
__global__ void k_ln(const float* __restrict__ src, float* __restrict__ dst,
                     const float* __restrict__ g, const float* __restrict__ b, int D) {
    __shared__ float red[256];
    const int tid = threadIdx.x;
    const float* x = src + (size_t)blockIdx.x * D;
    float* o = dst + (size_t)blockIdx.x * D;
    float s = 0.f;
    for (int i = tid; i < D; i += 256) s += x[i];
    red[tid] = s; __syncthreads();
    for (int off = 128; off > 0; off >>= 1) { if (tid < off) red[tid] += red[tid + off]; __syncthreads(); }
    const float mean = red[0] / D; __syncthreads();
    float v = 0.f;
    for (int i = tid; i < D; i += 256) { float d = x[i] - mean; v += d * d; }
    red[tid] = v; __syncthreads();
    for (int off = 128; off > 0; off >>= 1) { if (tid < off) red[tid] += red[tid + off]; __syncthreads(); }
    const float rstd = rsqrtf(red[0] / D + 1e-5f);
    __syncthreads();
    for (int i = tid; i < D; i += 256) o[i] = (x[i] - mean) * rstd * g[i] + b[i];
}

// ---------------------------------------------------------------------------
extern "C" void kernel_launch(void* const* d_in, const int* in_sizes, int n_in,
                              void* d_out, int out_size, void* d_ws, size_t ws_size,
                              hipStream_t stream) {
    const float* x        = (const float*)d_in[0];
    const float* lang_emb = (const float*)d_in[1];
    const float* hyper_w  = (const float*)d_in[2];
    const float* wih[2][2] = {{(const float*)d_in[3],  (const float*)d_in[7]},
                              {(const float*)d_in[13], (const float*)d_in[17]}};
    const float* whh[2][2] = {{(const float*)d_in[4],  (const float*)d_in[8]},
                              {(const float*)d_in[14], (const float*)d_in[18]}};
    const float* bih[2][2] = {{(const float*)d_in[5],  (const float*)d_in[9]},
                              {(const float*)d_in[15], (const float*)d_in[19]}};
    const float* bhh[2][2] = {{(const float*)d_in[6],  (const float*)d_in[10]},
                              {(const float*)d_in[16], (const float*)d_in[20]}};
    const float* lng[2] = {(const float*)d_in[11], (const float*)d_in[21]};
    const float* lnb[2] = {(const float*)d_in[12], (const float*)d_in[22]};
    const int*   lang_id = (const int*)d_in[23];

    float* out   = (float*)d_out;
    float* out_x = out;                                // (8,512,2048)
    float* out_h = out + (size_t)BB * TT * DMODEL;     // (2,8,1024)
    float* out_c = out_h + 2 * BB * HDIM;              // (2,8,1024)

    // ---- workspace carve-out ----
    char* ws = (char*)d_ws;
    size_t off = 0;
    auto alloc = [&](size_t bytes) -> void* {
        void* p = ws + off;
        off += (bytes + 255) & ~(size_t)255;
        return p;
    };
    float*  params  = (float*)alloc((size_t)2 * LAYERSZ * 4);
    __bf16* wihbf[2][2];
    wihbf[0][0] = (__bf16*)alloc((size_t)NGATE * 1024 * 2);
    wihbf[0][1] = (__bf16*)alloc((size_t)NGATE * 1024 * 2);
    wihbf[1][0] = (__bf16*)alloc((size_t)NGATE * 2048 * 2);
    wihbf[1][1] = (__bf16*)alloc((size_t)NGATE * 2048 * 2);
    __bf16* whhbf[2][2];
    whhbf[0][0] = (__bf16*)alloc((size_t)NGATE * HDIM * 2);
    whhbf[0][1] = (__bf16*)alloc((size_t)NGATE * HDIM * 2);
    whhbf[1][0] = (__bf16*)alloc((size_t)NGATE * HDIM * 2);
    whhbf[1][1] = (__bf16*)alloc((size_t)NGATE * HDIM * 2);
    __bf16* Abf   = (__bf16*)alloc((size_t)MROWS * 2048 * 2);
    float*  xgF   = (float*)alloc((size_t)MROWS * NGATE * 4);
    float*  xgR   = (float*)alloc((size_t)MROWS * NGATE * 4);
    float*  ybuf  = (float*)alloc((size_t)MROWS * DMODEL * 4);
    float*  x1    = (float*)alloc((size_t)MROWS * DMODEL * 4);
    float*  t1    = (float*)alloc((size_t)MROWS * DADAPT * 4);
    __bf16* t1bf  = (__bf16*)alloc((size_t)MROWS * DADAPT * 2);
    __bf16* wdbf  = (__bf16*)alloc((size_t)WSZ * 2);
    __bf16* wubf  = (__bf16*)alloc((size_t)WSZ * 2);
    __bf16* hbufF = (__bf16*)alloc((size_t)BB * HDIM * 2);
    __bf16* hbufR = (__bf16*)alloc((size_t)BB * HDIM * 2);
    unsigned* cnt = (unsigned*)alloc(256);
    (void)ws_size; (void)in_sizes; (void)n_in; (void)out_size;

    auto conv = [&](const float* s, __bf16* d, long long n) {
        k_f32_to_bf16<<<1024, 256, 0, stream>>>(s, d, n);
    };
    auto gemm = [&](const __bf16* A, const __bf16* Bm, float* C,
                    const float* b0, const float* b1, int M, int N, int K, int relu) {
        dim3 grid((N + 127) / 128, (M + 63) / 64);
        k_gemm_nt<<<grid, 256, 0, stream>>>(A, Bm, C, b0, b1, M, N, K, relu);
    };

    // hypernet: params = hyper_w @ lang_emb[lang_id]
    k_params<<<(2 * LAYERSZ + 255) / 256, 256, 0, stream>>>(hyper_w, lang_emb, lang_id,
                                                            params, 2 * LAYERSZ);
    // static weight conversion (w_hh in bf16 for TDM preload)
    conv(wih[0][0], wihbf[0][0], (long long)NGATE * 1024);
    conv(wih[0][1], wihbf[0][1], (long long)NGATE * 1024);
    conv(wih[1][0], wihbf[1][0], (long long)NGATE * 2048);
    conv(wih[1][1], wihbf[1][1], (long long)NGATE * 2048);
    conv(whh[0][0], whhbf[0][0], (long long)NGATE * HDIM);
    conv(whh[0][1], whhbf[0][1], (long long)NGATE * HDIM);
    conv(whh[1][0], whhbf[1][0], (long long)NGATE * HDIM);
    conv(whh[1][1], whhbf[1][1], (long long)NGATE * HDIM);

    const size_t rec_smem = 128 * 1024 * 2 + 16 * 1024 * 2 + 128 * 8 * 4 + 32 * 8 * 4; // 300032B

    for (int l = 0; l < 2; ++l) {
        const int Kin = (l == 0) ? 1024 : 2048;
        const float* xin = (l == 0) ? x : x1;

        // xg = x @ w_ih^T + (b_ih + b_hh), both directions
        conv(xin, Abf, (long long)MROWS * Kin);
        gemm(Abf, wihbf[l][0], xgF, bih[l][0], bhh[l][0], MROWS, NGATE, Kin, 0);
        gemm(Abf, wihbf[l][1], xgR, bih[l][1], bhh[l][1], MROWS, NGATE, Kin, 0);

        // sequential recurrence (persistent kernel, grid barriers)
        k_zero_u32<<<1, 64, 0, stream>>>(cnt, 64);
        k_recurrence<<<64, 256, rec_smem, stream>>>(
            xgF, xgR, whhbf[l][0], whhbf[l][1], ybuf, hbufF, hbufR,
            (l == 1) ? out_h : nullptr, (l == 1) ? out_c : nullptr, cnt);

        // adapter: relu(y @ w_down^T) @ w_up^T, then LayerNorm
        conv(params + (size_t)l * LAYERSZ, wdbf, WSZ);
        conv(params + (size_t)l * LAYERSZ + WSZ, wubf, WSZ);
        conv(ybuf, Abf, (long long)MROWS * DMODEL);
        gemm(Abf, wdbf, t1, nullptr, nullptr, MROWS, DADAPT, DMODEL, 1);
        conv(t1, t1bf, (long long)MROWS * DADAPT);
        float* dst = (l == 0) ? x1 : out_x;
        gemm(t1bf, wubf, dst, nullptr, nullptr, MROWS, DMODEL, DADAPT, 0);
        k_ln<<<MROWS, 256, 0, stream>>>(dst, dst, lng[l], lnb[l], DMODEL);
    }
}
